// TemporalEmbeddingModel_43757126811688
// MI455X (gfx1250) — compile-verified, matrix operand order checked by reading, not run on hardware
//
#include <hip/hip_runtime.h>
#include <math.h>

// ---------------------------------------------------------------------------
// Model dims
// ---------------------------------------------------------------------------
#define H_DIM   2048
#define B_DIM   32
#define S_DIM   512
#define NH_DIM  8
#define HD_DIM  256
#define NTOK    (B_DIM * S_DIM)   // 16384 tokens

typedef __attribute__((ext_vector_type(16))) __bf16 bf16x16;
typedef __attribute__((ext_vector_type(8)))  float  f32x8;
typedef int v4i __attribute__((vector_size(16)));   // 4 x i32, 16 B

// ---------------------------------------------------------------------------
// Async global->LDS copy (CDNA5 GLOBAL_LOAD_ASYNC_TO_LDS_B128, ASYNCcnt),
// with a plain-copy fallback if the toolchain lacks the builtins.
// Builtin signature (from clang diagnostic): (v4i AS1*, v4i AS3*, imm, imm).
// ---------------------------------------------------------------------------
#if defined(__has_builtin)
#if __has_builtin(__builtin_amdgcn_global_load_async_to_lds_b128) && \
    __has_builtin(__builtin_amdgcn_s_wait_asynccnt)
#define USE_ASYNC_LDS 1
#endif
#endif
#ifndef USE_ASYNC_LDS
#define USE_ASYNC_LDS 0
#endif

#if USE_ASYNC_LDS
#define ASYNC_CP16(dst_lds, src_glb)                                       \
  __builtin_amdgcn_global_load_async_to_lds_b128(                          \
      (__attribute__((address_space(1))) v4i*)(src_glb),                   \
      (__attribute__((address_space(3))) v4i*)(dst_lds), 0, 0)
#define ASYNC_WAIT() __builtin_amdgcn_s_wait_asynccnt(0)
#else
#define ASYNC_CP16(dst_lds, src_glb)                                       \
  (*(uint4*)(dst_lds) = *(const uint4*)(src_glb))
#define ASYNC_WAIT() ((void)0)
#endif

__device__ __forceinline__ unsigned short f2bf(float f) {
  unsigned u = __builtin_bit_cast(unsigned, f);
  u += 0x7fffu + ((u >> 16) & 1u);           // round-to-nearest-even
  return (unsigned short)(u >> 16);
}

// ---------------------------------------------------------------------------
// fp32 -> bf16 weight conversion (bf16 weights = 134 MB -> L2-resident)
// ---------------------------------------------------------------------------
__global__ __launch_bounds__(256)
void f32_to_bf16_k(const float* __restrict__ src,
                   unsigned short* __restrict__ dst, int n) {
  int i = blockIdx.x * 256 + threadIdx.x;
  if (i < n) dst[i] = f2bf(src[i]);
}

// ---------------------------------------------------------------------------
// x = embeddings + pe[temporal_indices]  -> bf16 activation for qkv GEMM
// ---------------------------------------------------------------------------
__global__ __launch_bounds__(256)
void pe_add_k(const float* __restrict__ emb, const int* __restrict__ tix,
              const float* __restrict__ pe, unsigned short* __restrict__ xbf) {
  size_t idx = (size_t)blockIdx.x * 256 + threadIdx.x;
  int tok = (int)(idx >> 11);          // / 2048
  int col = (int)(idx & 2047);
  int pos = tix[tok];
  xbf[idx] = f2bf(emb[idx] + pe[(size_t)pos * H_DIM + col]);
}

// ---------------------------------------------------------------------------
// Generic bf16 WMMA GEMM:  out = act( A[M,K] @ W[N,K]^T + bias (+ res) )
// Block tile 128x128, 8 waves (4x2), wave tile 32x64 = 2x4 WMMA frags.
// Double-buffered LDS staging via async copy; 8 WMMA per 32-wide k-step.
// ACT: 0=none 1=relu 2=gelu(exact).  Dual outputs: bf16 and/or fp32.
// ---------------------------------------------------------------------------
template <int ACT, bool RES>
__global__ __launch_bounds__(256)
void gemm_bf16_wmma(const unsigned short* __restrict__ A,
                    const unsigned short* __restrict__ W,
                    const float* __restrict__ bias,
                    const float* __restrict__ res,
                    unsigned short* __restrict__ obf,
                    float* __restrict__ of32,
                    int M, int N, int K) {
  __shared__ __align__(32) unsigned short As[2][128 * 32];  // 2 x 8 KB
  __shared__ __align__(32) unsigned short Ws[2][128 * 32];  // 2 x 8 KB

  const int tid  = threadIdx.x;
  const int lane = tid & 31;
  const int wid  = tid >> 5;
  const int wm   = wid >> 1;           // 0..3  (M sub-tile)
  const int wn   = wid & 1;            // 0..1  (N sub-tile, 64 wide)
  const int half = lane >> 4;
  const int l16  = lane & 15;
  const int m0   = blockIdx.y * 128;
  const int n0   = blockIdx.x * 128;

  f32x8 acc[2][4] = {};

  // stage one 128x32 A tile and 128x32 W tile into LDS buffer `buf`
  auto stage = [&](int buf, int k0) {
#pragma unroll
    for (int i = 0; i < 2; ++i) {
      int idx = tid * 2 + i;           // 512 chunks of 16B
      int r = idx >> 2, c = idx & 3;
      ASYNC_CP16(&As[buf][idx * 8], A + (size_t)(m0 + r) * K + k0 + c * 8);
      ASYNC_CP16(&Ws[buf][idx * 8], W + (size_t)(n0 + r) * K + k0 + c * 8);
    }
  };

  const int nk = K / 32;
  stage(0, 0);
  for (int t = 0; t < nk; ++t) {
    ASYNC_WAIT();
    __syncthreads();                   // tile `t&1` ready, prev reads done
    if (t + 1 < nk) stage((t + 1) & 1, (t + 1) * 32);
    const unsigned short* Ab = As[t & 1];
    const unsigned short* Wb = Ws[t & 1];

    bf16x16 a[2], b[4];
#pragma unroll
    for (int m = 0; m < 2; ++m)
      a[m] = *(const bf16x16*)(Ab + (wm * 32 + m * 16 + l16) * 32 + half * 16);
#pragma unroll
    for (int n = 0; n < 4; ++n)
      b[n] = *(const bf16x16*)(Wb + (wn * 64 + n * 16 + l16) * 32 + half * 16);

#pragma unroll
    for (int m = 0; m < 2; ++m)
#pragma unroll
      for (int n = 0; n < 4; ++n)
        acc[m][n] = __builtin_amdgcn_wmma_f32_16x16x32_bf16(
            false, a[m], false, b[n], (short)0, acc[m][n], false, false);
  }

  // Epilogue. C layout: VGPR i -> row i (lanes 0-15) / row 8+i (lanes 16-31)
#pragma unroll
  for (int m = 0; m < 2; ++m) {
#pragma unroll
    for (int n = 0; n < 4; ++n) {
      int col = n0 + wn * 64 + n * 16 + l16;
      int rb  = m0 + wm * 32 + m * 16 + half * 8;
      float bc = bias[col];
#pragma unroll
      for (int i = 0; i < 8; ++i) {
        size_t o = (size_t)(rb + i) * N + col;
        float v = acc[m][n][i] + bc;
        if (RES) v += res[o];
        if (ACT == 1) v = fmaxf(v, 0.0f);
        if (ACT == 2) v = 0.5f * v * (1.0f + erff(v * 0.70710678118654752f));
        if (of32) of32[o] = v;
        if (obf)  obf[o]  = f2bf(v);
      }
    }
  }
}

// ---------------------------------------------------------------------------
// Fused flash attention (bf16 WMMA, fp32 accum, online softmax).
// Grid: (S/128, NH, B). 8 waves; wave owns 16 q-rows end-to-end.
// k tile staged in LDS (async) so the s=qk^T loop is ds-fed, not
// global-load serialized; v staged transposed for the p@v WMMAs.
// ---------------------------------------------------------------------------
__global__ __launch_bounds__(256)
void attention_wmma(const unsigned short* __restrict__ qkv,
                    unsigned short* __restrict__ obf) {
  __shared__ __align__(32) unsigned short kS[64 * HD_DIM];   // 32 KB [key][d]
  __shared__ __align__(32) unsigned short vT[HD_DIM * 64];   // 32 KB [d][key]
  __shared__ __align__(32) unsigned short pst[8 * 16 * 64];  // 16 KB per-wave p

  const int qb = blockIdx.x, h = blockIdx.y, bb = blockIdx.z;
  const int tid = threadIdx.x, wid = tid >> 5, lane = tid & 31;
  const int half = lane >> 4, l16 = lane & 15;
  const size_t rs = 3 * H_DIM;  // 6144 elements per token row of packed qkv
  const unsigned short* base = qkv + (size_t)bb * S_DIM * rs;

  // Preload q fragments for this wave's 16 rows (K=256 -> 8 steps of 32)
  const int qrow = qb * 128 + wid * 16 + l16;
  const unsigned short* qp = base + (size_t)qrow * rs + h * HD_DIM;
  bf16x16 qf[8];
#pragma unroll
  for (int kk = 0; kk < 8; ++kk)
    qf[kk] = *(const bf16x16*)(qp + kk * 32 + half * 16);

  f32x8 o[16] = {};               // 16x256 fp32 accumulator per wave
  float mrow[8], lrow[8];
#pragma unroll
  for (int i = 0; i < 8; ++i) { mrow[i] = -1e30f; lrow[i] = 0.0f; }

  for (int j0 = 0; j0 < S_DIM; j0 += 64) {
    // Stage k tile [64 keys][256 d], async b128: 2048 chunks, 8/thread
#pragma unroll
    for (int i = 0; i < 8; ++i) {
      int idx = tid * 8 + i;
      int krow = idx >> 5, c = idx & 31;
      ASYNC_CP16(&kS[idx * 8],
                 base + (size_t)(j0 + krow) * rs + H_DIM + h * HD_DIM + c * 8);
    }
    // Stage v^T (64 keys x 256 d -> [d][key]): vector reads, b16 scatters
    {
      int key  = tid & 63;
      int dseg = tid >> 6;  // 0..3, 64 d-values each
      const unsigned short* vrow =
          base + (size_t)(j0 + key) * rs + 2 * H_DIM + h * HD_DIM + dseg * 64;
#pragma unroll
      for (int i = 0; i < 8; ++i) {
        uint4 t4 = ((const uint4*)vrow)[i];
        int db = dseg * 64 + i * 8;
        vT[(db + 0) * 64 + key] = (unsigned short)t4.x;
        vT[(db + 1) * 64 + key] = (unsigned short)(t4.x >> 16);
        vT[(db + 2) * 64 + key] = (unsigned short)t4.y;
        vT[(db + 3) * 64 + key] = (unsigned short)(t4.y >> 16);
        vT[(db + 4) * 64 + key] = (unsigned short)t4.z;
        vT[(db + 5) * 64 + key] = (unsigned short)(t4.z >> 16);
        vT[(db + 6) * 64 + key] = (unsigned short)t4.w;
        vT[(db + 7) * 64 + key] = (unsigned short)(t4.w >> 16);
      }
    }
    ASYNC_WAIT();
    __syncthreads();

    // s = q @ k^T, k fragments from LDS
    f32x8 s[4] = {};
#pragma unroll
    for (int n = 0; n < 4; ++n) {
#pragma unroll
      for (int kk = 0; kk < 8; ++kk) {
        bf16x16 kf = *(const bf16x16*)(kS + (n * 16 + l16) * HD_DIM +
                                       kk * 32 + half * 16);
        s[n] = __builtin_amdgcn_wmma_f32_16x16x32_bf16(
            false, qf[kk], false, kf, (short)0, s[n], false, false);
      }
    }

    // Online softmax. Lane-half owns rows half*8+i; reduce over 16 lanes.
    float corr[8];
#pragma unroll
    for (int i = 0; i < 8; ++i) {
      float mx = -1e30f;
#pragma unroll
      for (int n = 0; n < 4; ++n) {
        s[n][i] *= 0.0625f;               // 1/sqrt(HD)
        mx = fmaxf(mx, s[n][i]);
      }
#pragma unroll
      for (int off = 8; off; off >>= 1)
        mx = fmaxf(mx, __shfl_xor(mx, off, 16));
      float mn = fmaxf(mrow[i], mx);
      corr[i] = __expf(mrow[i] - mn);
      mrow[i] = mn;
      float rsum = 0.0f;
#pragma unroll
      for (int n = 0; n < 4; ++n) {
        float p = __expf(s[n][i] - mn);
        s[n][i] = p;
        rsum += p;
      }
#pragma unroll
      for (int off = 8; off; off >>= 1)
        rsum += __shfl_xor(rsum, off, 16);
      lrow[i] = lrow[i] * corr[i] + rsum;
    }
#pragma unroll
    for (int n = 0; n < 16; ++n)
#pragma unroll
      for (int i = 0; i < 8; ++i)
        o[n][i] *= corr[i];

    // p (C-layout) -> LDS as bf16, re-read as A-frags; o += p @ v
    unsigned short* pw = pst + wid * (16 * 64);
#pragma unroll
    for (int n = 0; n < 4; ++n)
#pragma unroll
      for (int i = 0; i < 8; ++i)
        pw[(half * 8 + i) * 64 + n * 16 + l16] = f2bf(s[n][i]);

#pragma unroll
    for (int kk2 = 0; kk2 < 2; ++kk2) {
      bf16x16 pf = *(const bf16x16*)(pw + l16 * 64 + kk2 * 32 + half * 16);
#pragma unroll
      for (int n = 0; n < 16; ++n) {
        bf16x16 vf =
            *(const bf16x16*)(vT + (n * 16 + l16) * 64 + kk2 * 32 + half * 16);
        o[n] = __builtin_amdgcn_wmma_f32_16x16x32_bf16(
            false, pf, false, vf, (short)0, o[n], false, false);
      }
    }
    __syncthreads();
  }

  // Normalize by l and emit bf16 for the out_proj GEMM
#pragma unroll
  for (int i = 0; i < 8; ++i) lrow[i] = 1.0f / fmaxf(lrow[i], 1e-20f);
#pragma unroll
  for (int n = 0; n < 16; ++n) {
#pragma unroll
    for (int i = 0; i < 8; ++i) {
      int row = qb * 128 + wid * 16 + half * 8 + i;
      int d   = n * 16 + l16;
      obf[((size_t)(bb * S_DIM + row)) * H_DIM + h * HD_DIM + d] =
          f2bf(o[n][i] * lrow[i]);
    }
  }
}

// ---------------------------------------------------------------------------
// LayerNorm per token row (2048 cols); optional fp32 out + bf16 out.
// ---------------------------------------------------------------------------
__global__ __launch_bounds__(256)
void layernorm_k(const float* __restrict__ X, const float* __restrict__ g,
                 const float* __restrict__ b, float* __restrict__ yf,
                 unsigned short* __restrict__ ybf) {
  __shared__ float sh[256];
  const int row = blockIdx.x, tid = threadIdx.x;
  const float* x = X + (size_t)row * H_DIM;
  float s = 0.0f, s2 = 0.0f;
  for (int c = tid; c < H_DIM; c += 256) {
    float v = x[c];
    s += v; s2 += v * v;
  }
  sh[tid] = s; __syncthreads();
  for (int o = 128; o; o >>= 1) { if (tid < o) sh[tid] += sh[tid + o]; __syncthreads(); }
  float mean = sh[0] * (1.0f / H_DIM); __syncthreads();
  sh[tid] = s2; __syncthreads();
  for (int o = 128; o; o >>= 1) { if (tid < o) sh[tid] += sh[tid + o]; __syncthreads(); }
  float var = sh[0] * (1.0f / H_DIM) - mean * mean;
  float inv = rsqrtf(var + 1e-5f);
  for (int c = tid; c < H_DIM; c += 256) {
    float v = (x[c] - mean) * inv * g[c] + b[c];
    if (yf) yf[(size_t)row * H_DIM + c] = v;
    ybf[(size_t)row * H_DIM + c] = f2bf(v);
  }
}

// ---------------------------------------------------------------------------
// out = x / max(||x||_2, 1e-12) per token row
// ---------------------------------------------------------------------------
__global__ __launch_bounds__(256)
void l2norm_k(const float* __restrict__ X, float* __restrict__ Y) {
  __shared__ float sh[256];
  const int row = blockIdx.x, tid = threadIdx.x;
  const float* x = X + (size_t)row * H_DIM;
  float s2 = 0.0f;
  for (int c = tid; c < H_DIM; c += 256) { float v = x[c]; s2 += v * v; }
  sh[tid] = s2; __syncthreads();
  for (int o = 128; o; o >>= 1) { if (tid < o) sh[tid] += sh[tid + o]; __syncthreads(); }
  float inv = 1.0f / fmaxf(sqrtf(sh[0]), 1e-12f);
  for (int c = tid; c < H_DIM; c += 256) Y[(size_t)row * H_DIM + c] = x[c] * inv;
}

// ---------------------------------------------------------------------------
// Orchestration
// ---------------------------------------------------------------------------
extern "C" void kernel_launch(void* const* d_in, const int* in_sizes, int n_in,
                              void* d_out, int out_size, void* d_ws,
                              size_t ws_size, hipStream_t stream) {
  (void)in_sizes; (void)n_in; (void)out_size; (void)ws_size;

  const float* emb   = (const float*)d_in[0];
  const int*   tix   = (const int*)d_in[1];
  const float* pe    = (const float*)d_in[2];
  const float* w_in  = (const float*)d_in[3];
  const float* b_in  = (const float*)d_in[4];
  const float* w_out = (const float*)d_in[5];
  const float* b_out = (const float*)d_in[6];
  const float* ln1g  = (const float*)d_in[7];
  const float* ln1b  = (const float*)d_in[8];
  const float* w_f1  = (const float*)d_in[9];
  const float* b_f1  = (const float*)d_in[10];
  const float* w_f2  = (const float*)d_in[11];
  const float* b_f2  = (const float*)d_in[12];
  const float* ln2g  = (const float*)d_in[13];
  const float* ln2b  = (const float*)d_in[14];
  const float* w_t1  = (const float*)d_in[15];
  const float* b_t1  = (const float*)d_in[16];
  const float* w_t2  = (const float*)d_in[17];
  const float* b_t2  = (const float*)d_in[18];
  const float* w_p1  = (const float*)d_in[19];
  const float* b_p1  = (const float*)d_in[20];
  const float* w_p2  = (const float*)d_in[21];
  const float* b_p2  = (const float*)d_in[22];
  float* out = (float*)d_out;

  // Workspace carve-out (256 B aligned slabs)
  char* ws = (char*)d_ws;
  size_t off = 0;
  auto take = [&](size_t bytes) -> char* {
    char* p = ws + off;
    off += (bytes + 255) & ~(size_t)255;
    return p;
  };
  const size_t HH = (size_t)H_DIM * H_DIM;
  const size_t TH = (size_t)NTOK * H_DIM;
  unsigned short* wb_in  = (unsigned short*)take(3 * HH * 2);
  unsigned short* wb_out = (unsigned short*)take(HH * 2);
  unsigned short* wb_f1  = (unsigned short*)take(4 * HH * 2);
  unsigned short* wb_f2  = (unsigned short*)take(4 * HH * 2);
  unsigned short* wb_t1  = (unsigned short*)take(HH * 2);
  unsigned short* wb_t2  = (unsigned short*)take(HH * 2);
  unsigned short* wb_p1  = (unsigned short*)take(HH * 2);
  unsigned short* wb_p2  = (unsigned short*)take(HH * 2);
  unsigned short* xbf    = (unsigned short*)take(TH * 2);
  unsigned short* qkvbf  = (unsigned short*)take(3 * TH * 2);
  unsigned short* atobf  = (unsigned short*)take(TH * 2);
  float*          r1     = (float*)take(TH * 4);
  float*          h1f    = (float*)take(TH * 4);
  unsigned short* h1bf   = (unsigned short*)take(TH * 2);
  unsigned short* f1bf   = (unsigned short*)take(4 * TH * 2);
  float*          r2     = (float*)take(TH * 4);
  unsigned short* h2bf   = (unsigned short*)take(TH * 2);
  unsigned short* t1bf   = (unsigned short*)take(TH * 2);
  unsigned short* tebf   = (unsigned short*)take(TH * 2);
  unsigned short* p1bf   = (unsigned short*)take(TH * 2);
  float*          prf    = (float*)take(TH * 4);

  dim3 blk(256);
  auto cvt = [&](const float* s, unsigned short* d, size_t n) {
    f32_to_bf16_k<<<dim3((unsigned)((n + 255) / 256)), blk, 0, stream>>>(s, d, (int)n);
  };
  cvt(w_in,  wb_in,  3 * HH);
  cvt(w_out, wb_out, HH);
  cvt(w_f1,  wb_f1,  4 * HH);
  cvt(w_f2,  wb_f2,  4 * HH);
  cvt(w_t1,  wb_t1,  HH);
  cvt(w_t2,  wb_t2,  HH);
  cvt(w_p1,  wb_p1,  HH);
  cvt(w_p2,  wb_p2,  HH);

  pe_add_k<<<dim3((unsigned)(TH / 256)), blk, 0, stream>>>(emb, tix, pe, xbf);

  auto G = [](int M, int N) { return dim3((unsigned)(N / 128), (unsigned)(M / 128)); };

  // qkv = x @ Win^T + b
  gemm_bf16_wmma<0, false><<<G(NTOK, 3 * H_DIM), blk, 0, stream>>>(
      xbf, wb_in, b_in, nullptr, qkvbf, nullptr, NTOK, 3 * H_DIM, H_DIM);
  // fused flash attention
  attention_wmma<<<dim3(S_DIM / 128, NH_DIM, B_DIM), blk, 0, stream>>>(qkvbf, atobf);
  // r1 = attn @ Wout^T + b + embeddings (residual fused)
  gemm_bf16_wmma<0, true><<<G(NTOK, H_DIM), blk, 0, stream>>>(
      atobf, wb_out, b_out, emb, nullptr, r1, NTOK, H_DIM, H_DIM);
  layernorm_k<<<NTOK, blk, 0, stream>>>(r1, ln1g, ln1b, h1f, h1bf);
  // f = gelu(h1 @ Wf1^T + b)
  gemm_bf16_wmma<2, false><<<G(NTOK, 4 * H_DIM), blk, 0, stream>>>(
      h1bf, wb_f1, b_f1, nullptr, f1bf, nullptr, NTOK, 4 * H_DIM, H_DIM);
  // r2 = f @ Wf2^T + b + h1 (residual fused)
  gemm_bf16_wmma<0, true><<<G(NTOK, H_DIM), blk, 0, stream>>>(
      f1bf, wb_f2, b_f2, h1f, nullptr, r2, NTOK, H_DIM, 4 * H_DIM);
  layernorm_k<<<NTOK, blk, 0, stream>>>(r2, ln2g, ln2b, nullptr, h2bf);
  // temporal encoder MLP
  gemm_bf16_wmma<1, false><<<G(NTOK, H_DIM), blk, 0, stream>>>(
      h2bf, wb_t1, b_t1, nullptr, t1bf, nullptr, NTOK, H_DIM, H_DIM);
  gemm_bf16_wmma<0, false><<<G(NTOK, H_DIM), blk, 0, stream>>>(
      t1bf, wb_t2, b_t2, nullptr, tebf, nullptr, NTOK, H_DIM, H_DIM);
  // projection head
  gemm_bf16_wmma<1, false><<<G(NTOK, H_DIM), blk, 0, stream>>>(
      tebf, wb_p1, b_p1, nullptr, p1bf, nullptr, NTOK, H_DIM, H_DIM);
  gemm_bf16_wmma<0, false><<<G(NTOK, H_DIM), blk, 0, stream>>>(
      p1bf, wb_p2, b_p2, nullptr, nullptr, prf, NTOK, H_DIM, H_DIM);
  // final L2 normalize
  l2norm_k<<<NTOK, blk, 0, stream>>>(prf, out);
}
